// HeteroGNN_GraphLevel_81552839016976
// MI455X (gfx1250) — compile-verified, matrix operand order checked by reading, not run on hardware
//
#include <hip/hip_runtime.h>
#include <hip/hip_bf16.h>
#include <stdint.h>

#define HD 128

typedef __attribute__((ext_vector_type(16))) __bf16 v16bf;
typedef __attribute__((ext_vector_type(8)))  float  v8f;

union BFr { v16bf v; unsigned short s[16]; uint4 q[2]; };

__device__ __forceinline__ unsigned short f2bf(float f) {
  unsigned u = __float_as_uint(f);
  u += 0x7FFFu + ((u >> 16) & 1u);      // round-to-nearest-even
  return (unsigned short)(u >> 16);
}

// ---------------------------------------------------------------------------
// Y[N x 128] = X[N x K] @ W[K x 128]   (K % 128 == 0)
// flags: 1 = add bias[col], 2 = accumulate into Y, 4 = relu
// bf16 WMMA (v_wmma_f32_16x16x32_bf16), fp32 accumulate.
// 256 threads = 8 waves; wave w computes rows [blk*128 + 16w, +16).
// ---------------------------------------------------------------------------
__global__ __launch_bounds__(256) void gemm_bf16_128(
    const float* __restrict__ X, const float* __restrict__ W,
    const float* __restrict__ bias, float* __restrict__ Y,
    int N, int K, int flags)
{
  __shared__ unsigned short ldsW[128 * 128];   // 32 KB, fragment-major bf16

  const int tid  = threadIdx.x;
  const int wave = tid >> 5, lane = tid & 31;
  const int g    = lane >> 4, lm = lane & 15;
  const int rowBase = blockIdx.x * 128 + wave * 16;

  v8f acc[8] = {};

  int row = rowBase + lm;
  if (row > N - 1) row = N - 1;                // clamp; OOB rows not written
  const float* xrow = X + (size_t)row * K;

  for (int k0 = 0; k0 < K; k0 += 128) {
    // Stage W chunk [k0, k0+128) x 128 into LDS in B-fragment order.
    for (int i = tid; i < 128 * 128; i += 256) {
      int frag = i >> 9, rem = i & 511;        // frag = kk*8 + t
      int kk = frag >> 3, t = frag & 7;
      int L = rem >> 4, e = rem & 15;
      int n = (t << 4) + (L & 15);
      int k = (kk << 5) + (e & 7) + ((L >> 4) << 3) + ((e >> 3) << 4);
      ldsW[i] = f2bf(W[(size_t)(k0 + k) * HD + n]);
    }
    __syncthreads();

#pragma unroll
    for (int kk = 0; kk < 4; kk++) {
      BFr a;
      {
        const float* pa = xrow + k0 + (kk << 5) + (g << 3);
        float4 f0 = *reinterpret_cast<const float4*>(pa);
        float4 f1 = *reinterpret_cast<const float4*>(pa + 4);
        float4 f2 = *reinterpret_cast<const float4*>(pa + 16);
        float4 f3 = *reinterpret_cast<const float4*>(pa + 20);
        a.s[0]=f2bf(f0.x); a.s[1]=f2bf(f0.y); a.s[2]=f2bf(f0.z); a.s[3]=f2bf(f0.w);
        a.s[4]=f2bf(f1.x); a.s[5]=f2bf(f1.y); a.s[6]=f2bf(f1.z); a.s[7]=f2bf(f1.w);
        a.s[8]=f2bf(f2.x); a.s[9]=f2bf(f2.y); a.s[10]=f2bf(f2.z); a.s[11]=f2bf(f2.w);
        a.s[12]=f2bf(f3.x); a.s[13]=f2bf(f3.y); a.s[14]=f2bf(f3.z); a.s[15]=f2bf(f3.w);
      }
#pragma unroll
      for (int t = 0; t < 8; t++) {
        BFr b;
        const uint4* q = reinterpret_cast<const uint4*>(
            &ldsW[(((kk << 3) + t) << 9) + (lane << 4)]);
        b.q[0] = q[0]; b.q[1] = q[1];
        acc[t] = __builtin_amdgcn_wmma_f32_16x16x32_bf16(
            false, a.v, false, b.v, (short)0, acc[t], false, false);
      }
    }
    __syncthreads();
  }

#pragma unroll
  for (int t = 0; t < 8; t++) {
    int col = (t << 4) + lm;
    float bv = (flags & 1) ? bias[col] : 0.f;
#pragma unroll
    for (int r = 0; r < 8; r++) {
      int orow = rowBase + (g << 3) + r;
      if (orow < N) {
        size_t o = (size_t)orow * HD + col;
        float v = acc[t][r] + bv;
        if (flags & 2) v += Y[o];
        if (flags & 4) v = fmaxf(v, 0.f);
        Y[o] = v;
      }
    }
  }
}

// --------------------------- elementwise / segment kernels ------------------

__global__ void k_count_dst(const int* __restrict__ d, float* cnt, int E) {
  int e = blockIdx.x * blockDim.x + threadIdx.x;
  if (e < E) atomicAdd(&cnt[d[e]], 1.0f);
}

__global__ void k_scatter_add(const float* __restrict__ X, const int* __restrict__ s,
                              const int* __restrict__ d, float* __restrict__ Y, int E) {
  long i = (long)blockIdx.x * blockDim.x + threadIdx.x;
  int e = (int)(i >> 7), f = (int)(i & 127);
  if (e < E) atomicAdd(&Y[(size_t)d[e] * HD + f], X[(size_t)s[e] * HD + f]);
}

__global__ void k_div_cnt(float* Y, const float* cnt, int N) {
  long i = (long)blockIdx.x * blockDim.x + threadIdx.x;
  if (i < (long)N * HD) Y[i] /= fmaxf(cnt[i >> 7], 1.f);
}

__global__ void k_rsqrt_deg(float* deg, int N) {   // deg -> dis = rsqrt(deg+1)
  int i = blockIdx.x * blockDim.x + threadIdx.x;
  if (i < N) deg[i] = rsqrtf(deg[i] + 1.0f);
}

__global__ void k_gcn_scatter(const float* __restrict__ Hf, const int* __restrict__ s,
                              const int* __restrict__ d, const float* __restrict__ dis,
                              float* __restrict__ Y, int E) {
  long i = (long)blockIdx.x * blockDim.x + threadIdx.x;
  int e = (int)(i >> 7), f = (int)(i & 127);
  if (e < E) {
    float w = dis[s[e]] * dis[d[e]];
    atomicAdd(&Y[(size_t)d[e] * HD + f], w * Hf[(size_t)s[e] * HD + f]);
  }
}

__global__ void k_gcn_finish(const float* __restrict__ Hf, const float* __restrict__ dis,
                             const float* __restrict__ b1, const float* __restrict__ b2,
                             float* __restrict__ Y, int N) {
  long i = (long)blockIdx.x * blockDim.x + threadIdx.x;
  if (i < (long)N * HD) {
    int n = (int)(i >> 7), f = (int)(i & 127);
    float dv = dis[n];
    float v = Y[i] + dv * dv * Hf[i] + b1[f];
    if (b2) v += b2[f];
    Y[i] = fmaxf(v, 0.f);
  }
}

__global__ __launch_bounds__(128) void k_dot_vec(const float* __restrict__ X,
                                                 const float* __restrict__ a,
                                                 float* __restrict__ out, int N) {
  int w = (int)(((long)blockIdx.x * blockDim.x + threadIdx.x) >> 5);
  int lane = threadIdx.x & 31;
  if (w >= N) return;
  const float* row = X + (size_t)w * HD;
  float s = 0.f;
#pragma unroll
  for (int j = 0; j < 4; j++) s += row[lane + 32 * j] * a[lane + 32 * j];
#pragma unroll
  for (int o = 16; o; o >>= 1) s += __shfl_xor(s, o, 32);
  if (lane == 0) out[w] = s;
}

__global__ void k_matvec(const float* __restrict__ W, const float* __restrict__ a,
                         float* __restrict__ out) {     // out[i] = sum_n W[i][n]*a[n]
  int i = threadIdx.x;
  float s = 0.f;
  for (int n = 0; n < HD; n++) s += W[(size_t)i * HD + n] * a[n];
  out[i] = s;
}

__device__ __forceinline__ unsigned keyenc(float f) {
  unsigned u = __float_as_uint(f);
  return (u & 0x80000000u) ? ~u : (u | 0x80000000u);
}
__device__ __forceinline__ float keydec(unsigned k) {
  unsigned u = (k & 0x80000000u) ? (k & 0x7FFFFFFFu) : ~k;
  return __uint_as_float(u);
}

__global__ void k_gat_logit(const int* __restrict__ s, const int* __restrict__ d,
                            const float* __restrict__ asd, const float* __restrict__ adv,
                            float* __restrict__ logit, unsigned* __restrict__ mb, int E) {
  int e = blockIdx.x * blockDim.x + threadIdx.x;
  if (e < E) {
    float l = asd[s[e]] + adv[d[e]];
    l = (l > 0.f) ? l : 0.2f * l;            // leaky_relu 0.2
    logit[e] = l;
    atomicMax(&mb[d[e]], keyenc(l));
  }
}

__global__ void k_gat_exp(const int* __restrict__ d, const float* __restrict__ logit,
                          const unsigned* __restrict__ mb, float* __restrict__ ebuf,
                          float* __restrict__ z, int E) {
  int e = blockIdx.x * blockDim.x + threadIdx.x;
  if (e < E) {
    float ev = expf(logit[e] - keydec(mb[d[e]]));
    ebuf[e] = ev;
    atomicAdd(&z[d[e]], ev);
  }
}

__global__ void k_gat_scatter(const int* __restrict__ s, const int* __restrict__ d,
                              const float* __restrict__ ebuf, const float* __restrict__ z,
                              const float* __restrict__ Hs, float* __restrict__ Y, int E) {
  long i = (long)blockIdx.x * blockDim.x + threadIdx.x;
  int e = (int)(i >> 7), f = (int)(i & 127);
  if (e < E) {
    float al = ebuf[e] / fmaxf(z[d[e]], 1e-16f);
    atomicAdd(&Y[(size_t)d[e] * HD + f], al * Hs[(size_t)s[e] * HD + f]);
  }
}

__global__ void k_bias_relu(float* __restrict__ Y, const float* __restrict__ b, long n) {
  long i = (long)blockIdx.x * blockDim.x + threadIdx.x;
  if (i < n) Y[i] = fmaxf(Y[i] + b[i & 127], 0.f);
}

__global__ void k_pool_sum(const float* __restrict__ X, const int* __restrict__ bidx,
                           float* __restrict__ emb, float* __restrict__ cnt,
                           int N, int colOff) {
  long i = (long)blockIdx.x * blockDim.x + threadIdx.x;
  int n = (int)(i >> 7), f = (int)(i & 127);
  if (n < N) {
    int g = bidx[n];
    atomicAdd(&emb[(size_t)g * 640 + colOff + f], X[i]);
    if (f == 0) atomicAdd(&cnt[g], 1.f);
  }
}

__global__ void k_pool_div(float* __restrict__ emb, const float* __restrict__ cnts, int G) {
  int i = blockIdx.x * blockDim.x + threadIdx.x;
  if (i < G * 640) {
    int g = i / 640, c = i - g * 640, set = c >> 7;
    emb[i] /= fmaxf(cnts[set * G + g], 1.f);
  }
}

__global__ __launch_bounds__(128) void k_lin2(const float* __restrict__ h1,
                                              const float* __restrict__ W2,
                                              const float* __restrict__ b2,
                                              float* __restrict__ out, int G) {
  int w = (int)(((long)blockIdx.x * blockDim.x + threadIdx.x) >> 5);
  int lane = threadIdx.x & 31;
  if (w >= G) return;
  float s = 0.f;
  for (int f = lane; f < HD; f += 32) s += h1[(size_t)w * HD + f] * W2[f];
#pragma unroll
  for (int o = 16; o; o >>= 1) s += __shfl_xor(s, o, 32);
  if (lane == 0) out[w] = s + b2[0];
}

// ------------------------------- host side ----------------------------------

static inline int cdiv(long a, int b) { return (int)((a + b - 1) / b); }

static void gemm(const float* X, const float* W, const float* b, float* Y,
                 int N, int K, int flags, hipStream_t st) {
  gemm_bf16_128<<<cdiv(N, 128), 256, 0, st>>>(X, W, b, Y, N, K, flags);
}

static void run_gat(const float* Xs, int Nsrc, const float* Xd, int Ndst,
                    const int* ei, int E,
                    const float* Ws, const float* Wd, const float* as, const float* ad,
                    float* Y, float* bufA, float* wvec, float* asd, float* adv,
                    unsigned* mb, float* z, float* elog, float* eexp, hipStream_t st) {
  const int* s = ei; const int* d = ei + E;
  gemm(Xs, Ws, nullptr, bufA, Nsrc, HD, 0, st);                 // Hs
  k_matvec<<<1, 128, 0, st>>>(Wd, ad, wvec);                    // Wd @ a_d
  k_dot_vec<<<cdiv((long)Nsrc * 32, 128), 128, 0, st>>>(bufA, as, asd, Nsrc);
  k_dot_vec<<<cdiv((long)Ndst * 32, 128), 128, 0, st>>>(Xd, wvec, adv, Ndst);
  (void)hipMemsetAsync(mb, 0, (size_t)Ndst * 4, st);
  (void)hipMemsetAsync(z, 0, (size_t)Ndst * 4, st);
  k_gat_logit<<<cdiv(E, 256), 256, 0, st>>>(s, d, asd, adv, elog, mb, E);
  k_gat_exp<<<cdiv(E, 256), 256, 0, st>>>(d, elog, mb, eexp, z, E);
  k_gat_scatter<<<cdiv((long)E * 128, 256), 256, 0, st>>>(s, d, eexp, z, bufA, Y, E);
}

static void run_gcn(const float* X, int N, const int* ei, int E, const float* W,
                    const float* b1, const float* b2, float* Y,
                    float* bufA, float* deg, hipStream_t st) {
  const int* s = ei; const int* d = ei + E;
  (void)hipMemsetAsync(deg, 0, (size_t)N * 4, st);
  k_count_dst<<<cdiv(E, 256), 256, 0, st>>>(d, deg, E);
  k_rsqrt_deg<<<cdiv(N, 256), 256, 0, st>>>(deg, N);
  gemm(X, W, nullptr, bufA, N, HD, 0, st);
  k_gcn_scatter<<<cdiv((long)E * 128, 256), 256, 0, st>>>(bufA, s, d, deg, Y, E);
  k_gcn_finish<<<cdiv((long)N * 128, 256), 256, 0, st>>>(bufA, deg, b1, b2, Y, N);
}

static void run_sage(const float* Xs, const float* Xd, int Ndst, const int* ei, int E,
                     const float* Wl, const float* bl, const float* Wr, float* Y,
                     float* agg, float* cnt, hipStream_t st) {
  const int* s = ei; const int* d = ei + E;
  (void)hipMemsetAsync(agg, 0, (size_t)Ndst * HD * 4, st);
  (void)hipMemsetAsync(cnt, 0, (size_t)Ndst * 4, st);
  k_scatter_add<<<cdiv((long)E * 128, 256), 256, 0, st>>>(Xs, s, d, agg, E);
  k_count_dst<<<cdiv(E, 256), 256, 0, st>>>(d, cnt, E);
  k_div_cnt<<<cdiv((long)Ndst * 128, 256), 256, 0, st>>>(agg, cnt, Ndst);
  gemm(agg, Wl, bl, Y, Ndst, HD, 1, st);            // mean @ Wl + bl
  gemm(Xd, Wr, nullptr, Y, Ndst, HD, 2 | 4, st);    // += x_dst @ Wr, relu
}

extern "C" void kernel_launch(void* const* d_in, const int* in_sizes, int n_in,
                              void* d_out, int out_size, void* d_ws, size_t ws_size,
                              hipStream_t stream) {
  const float* x_in[5] = {(const float*)d_in[0], (const float*)d_in[1],
                          (const float*)d_in[2], (const float*)d_in[3],
                          (const float*)d_in[4]};
  int Ns[5];
  for (int i = 0; i < 5; i++) Ns[i] = in_sizes[i] / HD;
  const int N_CD = Ns[0], N_SEQ = Ns[1], N_ANNO = Ns[2], N_RANGE = Ns[3], N_MD = Ns[4];

  const int* ei_seq  = (const int*)d_in[5];  const int E_seq  = in_sizes[5] / 2;
  const int* ei_anno = (const int*)d_in[6];  const int E_anno = in_sizes[6] / 2;
  const int* ei_rng  = (const int*)d_in[7];  const int E_rng  = in_sizes[7] / 2;
  const int* ei_comp = (const int*)d_in[8];  const int E_comp = in_sizes[8] / 2;
  const int* ei_cc   = (const int*)d_in[9];  const int E_cc   = in_sizes[9] / 2;
  const int* ei_mm   = (const int*)d_in[10]; const int E_mm   = in_sizes[10] / 2;
  const int* bidx[5] = {(const int*)d_in[11], (const int*)d_in[12], (const int*)d_in[13],
                        (const int*)d_in[14], (const int*)d_in[15]};

  const float* Wsl_seq  = (const float*)d_in[16]; const float* bs_seq  = (const float*)d_in[17];
  const float* Wsr_seq  = (const float*)d_in[18];
  const float* Wsl_anno = (const float*)d_in[19]; const float* bs_anno = (const float*)d_in[20];
  const float* Wsr_anno = (const float*)d_in[21];
  const float* Wg_cs = (const float*)d_in[22]; const float* Wg_cd = (const float*)d_in[23];
  const float* ag_cs = (const float*)d_in[24]; const float* ag_cd = (const float*)d_in[25];
  const float* bg_c  = (const float*)d_in[26];
  const float* Wg_rs = (const float*)d_in[27]; const float* Wg_rd = (const float*)d_in[28];
  const float* ag_rs = (const float*)d_in[29]; const float* ag_rd = (const float*)d_in[30];
  const float* bg_r  = (const float*)d_in[31];
  const float* Wc_cc = (const float*)d_in[32]; const float* bc_cc = (const float*)d_in[33];
  const float* Wc_mm = (const float*)d_in[34]; const float* bc_mm = (const float*)d_in[35];
  const float* lin1W = (const float*)d_in[36]; const float* lin1b = (const float*)d_in[37];
  const float* lin2W = (const float*)d_in[38]; const float* lin2b = (const float*)d_in[39];

  const int G = out_size;
  const size_t HH = (size_t)HD * HD;

  int EMAX = E_seq;
  if (E_anno > EMAX) EMAX = E_anno;
  if (E_rng  > EMAX) EMAX = E_rng;
  if (E_comp > EMAX) EMAX = E_comp;
  if (E_cc   > EMAX) EMAX = E_cc;
  if (E_mm   > EMAX) EMAX = E_mm;

  // workspace carve (all sizes multiples of 8 floats -> 32B alignment kept)
  float* base = (float*)d_ws; size_t off = 0;
  auto A = [&](size_t n) { float* p = base + off; off += n; return p; };
  float* set0[5]; float* set1[5];
  for (int i = 0; i < 5; i++) set0[i] = A((size_t)Ns[i] * HD);
  for (int i = 0; i < 5; i++) set1[i] = A((size_t)Ns[i] * HD);
  float* bufA = A((size_t)N_CD * HD);
  float* wvec = A(128);
  float* asd  = A((size_t)N_CD);
  float* adv  = A((size_t)N_CD);
  float* deg  = A((size_t)N_CD);
  float* zv   = A((size_t)N_CD);
  unsigned* mb = (unsigned*)A((size_t)N_CD);
  float* elog = A((size_t)EMAX);
  float* eexp = A((size_t)EMAX);
  float* emb  = A((size_t)G * 640);
  float* cntG = A((size_t)5 * G);
  float* h1   = A((size_t)G * HD);
  (void)ws_size; (void)n_in;

  for (int l = 0; l < 2; l++) {
    const float* Xc = l ? set0[0] : x_in[0];
    const float* Xs = l ? set0[1] : x_in[1];
    const float* Xa = l ? set0[2] : x_in[2];
    const float* Xr = l ? set0[3] : x_in[3];
    const float* Xm = l ? set0[4] : x_in[4];
    float** Yo = l ? set1 : set0;

    // CD: GAT(MD->CD) + GCN(CD->CD), biases + relu in gcn_finish
    (void)hipMemsetAsync(Yo[0], 0, (size_t)N_CD * HD * 4, stream);
    run_gat(Xm, N_MD, Xc, N_CD, ei_comp, E_comp,
            Wg_cs + l * HH, Wg_cd + l * HH, ag_cs + l * HD, ag_cd + l * HD,
            Yo[0], bufA, wvec, asd, adv, mb, zv, elog, eexp, stream);
    run_gcn(Xc, N_CD, ei_cc, E_cc, Wc_cc + l * HH,
            bc_cc + l * HD, bg_c + l * HD, Yo[0], bufA, deg, stream);

    // SEQ / ANNO: SAGE(CD->*)
    run_sage(Xc, Xs, N_SEQ, ei_seq, E_seq,
             Wsl_seq + l * HH, bs_seq + l * HD, Wsr_seq + l * HH,
             Yo[1], bufA, deg, stream);
    run_sage(Xc, Xa, N_ANNO, ei_anno, E_anno,
             Wsl_anno + l * HH, bs_anno + l * HD, Wsr_anno + l * HH,
             Yo[2], bufA, deg, stream);

    // RANGE: GAT(CD->RANGE) + bias + relu
    (void)hipMemsetAsync(Yo[3], 0, (size_t)N_RANGE * HD * 4, stream);
    run_gat(Xc, N_CD, Xr, N_RANGE, ei_rng, E_rng,
            Wg_rs + l * HH, Wg_rd + l * HH, ag_rs + l * HD, ag_rd + l * HD,
            Yo[3], bufA, wvec, asd, adv, mb, zv, elog, eexp, stream);
    k_bias_relu<<<cdiv((long)N_RANGE * HD, 256), 256, 0, stream>>>(
        Yo[3], bg_r + l * HD, (long)N_RANGE * HD);

    // MD: GCN(MD->MD)
    (void)hipMemsetAsync(Yo[4], 0, (size_t)N_MD * HD * 4, stream);
    run_gcn(Xm, N_MD, ei_mm, E_mm, Wc_mm + l * HH,
            bc_mm + l * HD, nullptr, Yo[4], bufA, deg, stream);
  }

  // mean-pool each node set into emb[G x 640]
  (void)hipMemsetAsync(emb, 0, (size_t)G * 640 * 4, stream);
  (void)hipMemsetAsync(cntG, 0, (size_t)5 * G * 4, stream);
  for (int i = 0; i < 5; i++)
    k_pool_sum<<<cdiv((long)Ns[i] * HD, 256), 256, 0, stream>>>(
        set1[i], bidx[i], emb, cntG + (size_t)i * G, Ns[i], i * HD);
  k_pool_div<<<cdiv((long)G * 640, 256), 256, 0, stream>>>(emb, cntG, G);

  // MLP head: h1 = relu(emb @ lin1W + lin1b);  out = h1 @ lin2W + lin2b
  gemm(emb, lin1W, lin1b, h1, G, 640, 1 | 4, stream);
  k_lin2<<<cdiv((long)G * 32, 128), 128, 0, stream>>>(
      h1, lin2W, lin2b, (float*)d_out, G);
}
